// MultiHeadAttention_38491496907078
// MI455X (gfx1250) — matrix operand order repaired and checked by reading, not compiled, not run on hardware
//
#include <hip/hip_runtime.h>

// Problem constants
#define BB   4
#define TT   2048
#define CC   1024
#define HH   16
#define DD   64
#define N3   3072          // 3*C
#define MTOT 8192          // B*T
#define ATT_SCALE 0.125f   // 1/sqrt(64)

typedef __attribute__((ext_vector_type(16))) __bf16 v16bf;
typedef __attribute__((ext_vector_type(8)))  float  v8f;
typedef int v4i __attribute__((vector_size(16)));

struct alignas(8)  US4  { unsigned short s[4];  };
struct alignas(16) US8  { unsigned short s[8];  };
struct alignas(32) US16 { unsigned short s[16]; };

union BFrag {
    v16bf v;
    unsigned short u[16];
};

#if __has_builtin(__builtin_amdgcn_global_load_async_to_lds_b128)
#define HAVE_ASYNC_LDS 1
#endif

typedef __attribute__((address_space(1))) v4i* gp128_t;
typedef __attribute__((address_space(3))) v4i* lp128_t;

__device__ __forceinline__ void async_copy16(const unsigned short* g, unsigned short* l) {
#ifdef HAVE_ASYNC_LDS
    __builtin_amdgcn_global_load_async_to_lds_b128((gp128_t)g, (lp128_t)l, 0, 0);
#else
    *(US8*)l = *(const US8*)g;   // fallback: synchronous copy through VGPRs
#endif
}

template <int N>
__device__ __forceinline__ void wait_async() {
#ifdef HAVE_ASYNC_LDS
#if __has_builtin(__builtin_amdgcn_s_wait_asynccnt)
    __builtin_amdgcn_s_wait_asynccnt(N);
#else
    asm volatile("s_wait_asynccnt %0" :: "i"(N) : "memory");
#endif
#endif
}

__device__ __forceinline__ unsigned short f2bf(float f) {
    unsigned int u = __float_as_uint(f);
    u += 0x7fffu + ((u >> 16) & 1u);   // round to nearest even
    return (unsigned short)(u >> 16);
}

// Stage one 64col x 32K bf16 B-slice into LDS (4 KB) with one b128 per thread.
__device__ __forceinline__ void stageB(const unsigned short* __restrict__ wT,
                                       unsigned short* lds, int colbase, int kk, int tid) {
    int col   = tid >> 2;     // 0..63
    int chunk = tid & 3;      // 0..3 (8 bf16 each)
    const unsigned short* g = wT + (size_t)(colbase + col) * CC + kk + chunk * 8;
    async_copy16(g, lds + col * 32 + chunk * 8);
}

// ---------------------------------------------------------------------------
// Prep: fp32 -> bf16 (vectorized x4)
__global__ void cvt_f32_to_bf16(const float* __restrict__ in,
                                unsigned short* __restrict__ out, int n4) {
    int i = blockIdx.x * blockDim.x + threadIdx.x;
    if (i >= n4) return;
    float4 f = ((const float4*)in)[i];
    US4 o;
    o.s[0] = f2bf(f.x); o.s[1] = f2bf(f.y);
    o.s[2] = f2bf(f.z); o.s[3] = f2bf(f.w);
    ((US4*)out)[i] = o;
}

// Prep: W[K=1024][N] fp32 -> Wt[N][1024] bf16
__global__ void cvt_transposeW(const float* __restrict__ in,
                               unsigned short* __restrict__ out, int N) {
    long i = (long)blockIdx.x * blockDim.x + threadIdx.x;
    if (i >= (long)N * CC) return;
    int n = (int)(i >> 10);
    int k = (int)(i & 1023);
    out[i] = f2bf(in[(long)k * N + n]);
}

// ---------------------------------------------------------------------------
// GEMM 1: qkv = xb[8192,1024] @ WqkvT^T.  Block = 256 threads = 8 waves,
// block tile 256(M)x64(N); per-wave 32x64 (2 A-frag groups x 4 col tiles).
// B K-slices staged in LDS via async global->LDS, double buffered.
__global__ void __launch_bounds__(256)
gemm_qkv(const unsigned short* __restrict__ xb,
         const unsigned short* __restrict__ wT,
         unsigned short* __restrict__ qh,
         unsigned short* __restrict__ kh,
         unsigned short* __restrict__ vT) {
    __shared__ unsigned short bs[2][64][32];   // 8 KB, double buffered

    const int tid  = threadIdx.x;
    const int lane = tid & 31;
    const int wave = tid >> 5;
    const int l  = lane & 15;
    const int hf = lane >> 4;
    const int rowbase = blockIdx.x * 256 + wave * 32;
    const int colbase = blockIdx.y * 64;

    const unsigned short* Arow0 = xb + (size_t)(rowbase + l) * CC;
    const unsigned short* Arow1 = xb + (size_t)(rowbase + 16 + l) * CC;

    v8f z = {0.f,0.f,0.f,0.f,0.f,0.f,0.f,0.f};
    v8f acc[2][4] = {{z,z,z,z},{z,z,z,z}};

    stageB(wT, &bs[0][0][0], colbase, 0, tid);
    int buf = 0;
    for (int kk = 0; kk < CC; kk += 32) {
        if (kk + 32 < CC) { stageB(wT, &bs[buf ^ 1][0][0], colbase, kk + 32, tid); wait_async<1>(); }
        else              { wait_async<0>(); }
        __syncthreads();

        BFrag a[2];
        {
            US8 a00 = *(const US8*)(Arow0 + kk + hf * 16);
            US8 a01 = *(const US8*)(Arow0 + kk + hf * 16 + 8);
            US8 a10 = *(const US8*)(Arow1 + kk + hf * 16);
            US8 a11 = *(const US8*)(Arow1 + kk + hf * 16 + 8);
            #pragma unroll
            for (int i = 0; i < 8; ++i) {
                a[0].u[i] = a00.s[i]; a[0].u[8 + i] = a01.s[i];
                a[1].u[i] = a10.s[i]; a[1].u[8 + i] = a11.s[i];
            }
        }
        #pragma unroll
        for (int j = 0; j < 4; ++j) {
            BFrag b;
            US16 bv = *(const US16*)(&bs[buf][j * 16 + l][hf * 16]);
            #pragma unroll
            for (int i = 0; i < 16; ++i) b.u[i] = bv.s[i];
            acc[0][j] = __builtin_amdgcn_wmma_f32_16x16x32_bf16(
                false, a[0].v, false, b.v, (short)0, acc[0][j], false, false);
            acc[1][j] = __builtin_amdgcn_wmma_f32_16x16x32_bf16(
                false, a[1].v, false, b.v, (short)0, acc[1][j], false, false);
        }
        __syncthreads();
        buf ^= 1;
    }

    const int part = colbase >> 10;          // 0=q 1=k 2=v
    const int head = (colbase & 1023) >> 6;
    const int b    = rowbase >> 11;
    const int bh   = b * HH + head;

    #pragma unroll
    for (int g = 0; g < 2; ++g) {
        const int t0 = (rowbase + g * 16) & 2047;
        if (part < 2) {
            unsigned short* dst = (part == 0) ? qh : kh;
            #pragma unroll
            for (int j = 0; j < 4; ++j) {
                int d = j * 16 + l;
                #pragma unroll
                for (int r = 0; r < 8; ++r) {
                    int t = t0 + hf * 8 + r;
                    dst[((size_t)bh * TT + t) * DD + d] = f2bf(acc[g][j][r]);
                }
            }
        } else {
            // V transposed: [b,h,d,t]; 8 consecutive t per lane -> one b128 store
            #pragma unroll
            for (int j = 0; j < 4; ++j) {
                int d = j * 16 + l;
                US8 pk;
                #pragma unroll
                for (int r = 0; r < 8; ++r) pk.s[r] = f2bf(acc[g][j][r]);
                *(US8*)(vT + ((size_t)bh * DD + d) * TT + t0 + hf * 8) = pk;
            }
        }
    }
}

// ---------------------------------------------------------------------------
// Flash attention: one wave handles 16 queries of one (b,h); online softmax
// over 32-key blocks; S and P*V via bf16 WMMA; S->P transpose through LDS.
__global__ void __launch_bounds__(32)
attn_kernel(const unsigned short* __restrict__ qh,
            const unsigned short* __restrict__ kh,
            const unsigned short* __restrict__ vT,
            unsigned short* __restrict__ attnOut) {
    __shared__ unsigned short p_lds[16][32];

    const int lane  = threadIdx.x & 31;
    const int l     = lane & 15;
    const int hf    = lane >> 4;
    const int bh    = blockIdx.x;
    const int qbase = blockIdx.y * 16;

    const unsigned short* Q = qh + ((size_t)bh * TT + qbase) * DD;
    BFrag aq[2];
    #pragma unroll
    for (int c = 0; c < 2; ++c) {
        US8 a0 = *(const US8*)(Q + (size_t)l * DD + c * 32 + hf * 16);
        US8 a1 = *(const US8*)(Q + (size_t)l * DD + c * 32 + hf * 16 + 8);
        #pragma unroll
        for (int i = 0; i < 8; ++i) { aq[c].u[i] = a0.s[i]; aq[c].u[8 + i] = a1.s[i]; }
    }

    float m8[8], l8[8];
    #pragma unroll
    for (int r = 0; r < 8; ++r) { m8[r] = -1e30f; l8[r] = 0.f; }
    v8f z = {0.f,0.f,0.f,0.f,0.f,0.f,0.f,0.f};
    v8f acc[4] = {z, z, z, z};

    const int qlast = qbase + 15;
    for (int kk = 0; kk <= qlast; kk += 32) {
        v8f s[2] = {z, z};
        #pragma unroll
        for (int tile = 0; tile < 2; ++tile) {
            #pragma unroll
            for (int c = 0; c < 2; ++c) {
                BFrag bk;
                US16 kv = *(const US16*)(kh + ((size_t)bh * TT + kk + tile * 16 + l) * DD
                                            + c * 32 + hf * 16);
                #pragma unroll
                for (int i = 0; i < 16; ++i) bk.u[i] = kv.s[i];
                s[tile] = __builtin_amdgcn_wmma_f32_16x16x32_bf16(
                    false, aq[c].v, false, bk.v, (short)0, s[tile], false, false);
            }
        }

        const int key0 = kk + l;
        const int key1 = kk + 16 + l;
        float p0[8], p1[8];
        #pragma unroll
        for (int r = 0; r < 8; ++r) {
            int q = qbase + hf * 8 + r;
            float s0 = s[0][r] * ATT_SCALE; if (key0 > q) s0 = -1e30f;
            float s1 = s[1][r] * ATT_SCALE; if (key1 > q) s1 = -1e30f;
            float mx = fmaxf(s0, s1);
            #pragma unroll
            for (int sh = 1; sh < 16; sh <<= 1) mx = fmaxf(mx, __shfl_xor(mx, sh, 32));
            float mn = fmaxf(m8[r], mx);
            float e0 = __expf(s0 - mn);
            float e1 = __expf(s1 - mn);
            float sum = e0 + e1;
            #pragma unroll
            for (int sh = 1; sh < 16; sh <<= 1) sum += __shfl_xor(sum, sh, 32);
            float alpha = __expf(m8[r] - mn);
            l8[r] = l8[r] * alpha + sum;
            m8[r] = mn;
            p0[r] = e0; p1[r] = e1;
            #pragma unroll
            for (int j = 0; j < 4; ++j) acc[j][r] *= alpha;
        }

        __syncthreads();
        #pragma unroll
        for (int r = 0; r < 8; ++r) {
            p_lds[hf * 8 + r][l]      = f2bf(p0[r]);
            p_lds[hf * 8 + r][16 + l] = f2bf(p1[r]);
        }
        __syncthreads();
        BFrag ap;
        US8 pl0 = *(const US8*)(&p_lds[l][hf * 16]);
        US8 pl1 = *(const US8*)(&p_lds[l][hf * 16 + 8]);
        #pragma unroll
        for (int i = 0; i < 8; ++i) { ap.u[i] = pl0.s[i]; ap.u[8 + i] = pl1.s[i]; }

        #pragma unroll
        for (int j = 0; j < 4; ++j) {
            BFrag bv;
            US16 vv = *(const US16*)(vT + ((size_t)bh * DD + j * 16 + l) * TT
                                        + kk + hf * 16);
            #pragma unroll
            for (int i = 0; i < 16; ++i) bv.u[i] = vv.s[i];
            acc[j] = __builtin_amdgcn_wmma_f32_16x16x32_bf16(
                false, ap.v, false, bv.v, (short)0, acc[j], false, false);
        }
    }

    const int bidx = bh >> 4, hidx = bh & 15;
    #pragma unroll
    for (int j = 0; j < 4; ++j) {
        #pragma unroll
        for (int r = 0; r < 8; ++r) {
            int t = qbase + hf * 8 + r;
            float o = acc[j][r] / l8[r];
            attnOut[((size_t)(bidx * TT + t)) * CC + hidx * 64 + j * 16 + l] = f2bf(o);
        }
    }
}

// ---------------------------------------------------------------------------
// GEMM 3: out = attn[8192,1024] @ Wo + bo, fp32 output. Same scheme as GEMM 1.
__global__ void __launch_bounds__(256)
gemm_out(const unsigned short* __restrict__ attn,
         const unsigned short* __restrict__ woT,
         const float* __restrict__ bo,
         float* __restrict__ out) {
    __shared__ unsigned short bs[2][64][32];

    const int tid  = threadIdx.x;
    const int lane = tid & 31;
    const int wave = tid >> 5;
    const int l  = lane & 15;
    const int hf = lane >> 4;
    const int rowbase = blockIdx.x * 256 + wave * 32;
    const int colbase = blockIdx.y * 64;

    const unsigned short* Arow0 = attn + (size_t)(rowbase + l) * CC;
    const unsigned short* Arow1 = attn + (size_t)(rowbase + 16 + l) * CC;

    v8f z = {0.f,0.f,0.f,0.f,0.f,0.f,0.f,0.f};
    v8f acc[2][4] = {{z,z,z,z},{z,z,z,z}};

    stageB(woT, &bs[0][0][0], colbase, 0, tid);
    int buf = 0;
    for (int kk = 0; kk < CC; kk += 32) {
        if (kk + 32 < CC) { stageB(woT, &bs[buf ^ 1][0][0], colbase, kk + 32, tid); wait_async<1>(); }
        else              { wait_async<0>(); }
        __syncthreads();

        BFrag a[2];
        {
            US8 a00 = *(const US8*)(Arow0 + kk + hf * 16);
            US8 a01 = *(const US8*)(Arow0 + kk + hf * 16 + 8);
            US8 a10 = *(const US8*)(Arow1 + kk + hf * 16);
            US8 a11 = *(const US8*)(Arow1 + kk + hf * 16 + 8);
            #pragma unroll
            for (int i = 0; i < 8; ++i) {
                a[0].u[i] = a00.s[i]; a[0].u[8 + i] = a01.s[i];
                a[1].u[i] = a10.s[i]; a[1].u[8 + i] = a11.s[i];
            }
        }
        #pragma unroll
        for (int j = 0; j < 4; ++j) {
            BFrag b;
            US16 bv = *(const US16*)(&bs[buf][j * 16 + l][hf * 16]);
            #pragma unroll
            for (int i = 0; i < 16; ++i) b.u[i] = bv.s[i];
            acc[0][j] = __builtin_amdgcn_wmma_f32_16x16x32_bf16(
                false, a[0].v, false, b.v, (short)0, acc[0][j], false, false);
            acc[1][j] = __builtin_amdgcn_wmma_f32_16x16x32_bf16(
                false, a[1].v, false, b.v, (short)0, acc[1][j], false, false);
        }
        __syncthreads();
        buf ^= 1;
    }

    #pragma unroll
    for (int g = 0; g < 2; ++g) {
        #pragma unroll
        for (int j = 0; j < 4; ++j) {
            int col = colbase + j * 16 + l;
            float bias = bo[col];
            #pragma unroll
            for (int r = 0; r < 8; ++r) {
                int row = rowbase + g * 16 + hf * 8 + r;
                out[(size_t)row * CC + col] = acc[g][j][r] + bias;
            }
        }
    }
}

// ---------------------------------------------------------------------------
extern "C" void kernel_launch(void* const* d_in, const int* in_sizes, int n_in,
                              void* d_out, int out_size, void* d_ws, size_t ws_size,
                              hipStream_t stream) {
    (void)in_sizes; (void)n_in; (void)out_size; (void)ws_size;
    const float* x    = (const float*)d_in[0];
    // d_in[1] = causal mask (implicit in kernel)
    const float* Wqkv = (const float*)d_in[2];
    const float* Wo   = (const float*)d_in[3];
    const float* bo   = (const float*)d_in[4];
    float* out = (float*)d_out;

    char* ws = (char*)d_ws;
    unsigned short* xb    = (unsigned short*)(ws);              // 16 MB
    unsigned short* wqkvT = (unsigned short*)(ws + 16777216);   //  6 MB
    unsigned short* woT   = (unsigned short*)(ws + 23068672);   //  2 MB
    unsigned short* qh    = (unsigned short*)(ws + 25165824);   // 16 MB [b,h,t,d]
    unsigned short* kh    = (unsigned short*)(ws + 41943040);   // 16 MB [b,h,t,d]
    unsigned short* vT    = (unsigned short*)(ws + 58720256);   // 16 MB [b,h,d,t]
    unsigned short* attn  = (unsigned short*)(ws + 75497472);   // 16 MB

    cvt_f32_to_bf16<<<(MTOT * CC / 4 + 255) / 256, 256, 0, stream>>>(x, xb, MTOT * CC / 4);
    cvt_transposeW<<<(N3 * CC + 255) / 256, 256, 0, stream>>>(Wqkv, wqkvT, N3);
    cvt_transposeW<<<(CC * CC + 255) / 256, 256, 0, stream>>>(Wo, woT, CC);

    gemm_qkv<<<dim3(MTOT / 256, N3 / 64), 256, 0, stream>>>(xb, wqkvT, qh, kh, vT);
    attn_kernel<<<dim3(BB * HH, TT / 16), 32, 0, stream>>>(qh, kh, vT, attn);
    gemm_out<<<dim3(MTOT / 256, CC / 64), 256, 0, stream>>>(attn, woT, bo, out);
}
